// Decoder_9028021256859
// MI455X (gfx1250) — matrix operand order repaired and checked by reading, not compiled, not run on hardware
//
#include <hip/hip_runtime.h>
#include <hip/hip_bf16.h>
#include <math.h>

// ---------------- problem constants (match reference) ----------------
#define S_   56
#define B_   32
#define C_   1024
#define T_   48
#define H_   1024
#define EMB_ 512
#define RANK_ 128
#define VOCAB_ 32000
#define KX0_ (EMB_ + C_ + H_)   // 2560 : [emb | ctx | h0_prev] for fused W0;U0
#define KX1_ (2 * H_)           // 2048 : [h0_new | h1_prev]  for fused W1;U1
#define KF_  (H_ + C_)          // 2048 : feats = [h1 | ctx]
#define MOUT_ ((T_ - 1) * B_)   // 1504 rows (multiple of 16)

typedef __attribute__((ext_vector_type(16))) __bf16      v16bf;
typedef __attribute__((ext_vector_type(8)))  float       v8f;
typedef __attribute__((ext_vector_type(4)))  unsigned int v4u;

static __device__ __forceinline__ unsigned short f2bf(float f) {
    unsigned int u = __float_as_uint(f);
    unsigned int r = (u + 0x7FFFu + ((u >> 16) & 1u)) >> 16;   // RNE
    return (unsigned short)r;
}
static __device__ __forceinline__ float sigmoidf_(float x) {
    return 1.0f / (1.0f + __expf(-x));
}

// ---------------------------------------------------------------------
// Pack an f32 weight matrix (logical [K,N], or [N,K] if transposed) into
// the per-lane WMMA bf16 B-operand layout:
//   tile (nt,kt) -> 32 lanes x 16 contiguous bf16
//   lane l<16  : col n = nt*16+l,    K = {kt*32 + 0..7,  kt*32 + 16..23}
//   lane l>=16 : col n = nt*16+l-16, K = {kt*32 + 8..15, kt*32 + 24..31}
// kt_off lets us stack two weights (W;U) along K into one packed buffer.
// ---------------------------------------------------------------------
__global__ void pack_b_kernel(const float* __restrict__ src,
                              unsigned short* __restrict__ dst,
                              int N, int K, int KT_total, int kt_off,
                              int transposed)
{
    const int KTsrc = K >> 5;
    const long total = (long)(N >> 4) * KTsrc * 512;
    long idx = (long)blockIdx.x * blockDim.x + threadIdx.x;
    if (idx >= total) return;
    int within = (int)(idx & 511);
    int lane = within >> 4;
    int j    = within & 15;
    long tile = idx >> 9;
    int kt = (int)(tile % KTsrc);
    int nt = (int)(tile / KTsrc);
    int n = nt * 16 + (lane & 15);
    int k = kt * 32 + (j < 8 ? j : j + 8) + (lane < 16 ? 0 : 8);
    float v = transposed ? src[(size_t)n * K + k] : src[(size_t)k * N + n];
    dst[((size_t)(nt * KT_total + kt + kt_off) * 32 + lane) * 16 + j] = f2bf(v);
}

// ---------------------------------------------------------------------
// WMMA GEMM: C[M,N] = act(A_bf16[M,K] @ Bpacked + bias)
// block = 128 threads = 4 waves; each wave owns one 16x16 C tile.
// grid = (N/64, M/16).  Inner loop: 2x b128 A loads + 2x b128 B loads +
// one v_wmma_f32_16x16x32_bf16 per K-tile.
// ---------------------------------------------------------------------
__global__ void wmma_gemm_kernel(const unsigned short* __restrict__ A,
                                 const unsigned short* __restrict__ Bp,
                                 const float* __restrict__ bias,
                                 float* __restrict__ outF,
                                 unsigned short* __restrict__ outB,
                                 int M, int N, int K, int act)
{
    const int lane  = threadIdx.x & 31;
    const int wave  = threadIdx.x >> 5;
    const int ntile = blockIdx.x * 4 + wave;
    const int mtile = blockIdx.y;
    const int KT    = K >> 5;

    const int arow  = (lane & 15);
    const int base0 = (lane < 16) ? 0 : 8;
    const unsigned short* Arow = A + ((size_t)(mtile * 16 + arow)) * K + base0;
    const unsigned short* Bl   = Bp + ((size_t)ntile * KT * 32 + lane) * 16;

    v8f acc = {0.f, 0.f, 0.f, 0.f, 0.f, 0.f, 0.f, 0.f};

    for (int kt = 0; kt < KT; ++kt) {
        union { v4u u[2]; v16bf v; } ua, ub;
        const unsigned short* ap = Arow + kt * 32;
        ua.u[0] = *(const v4u*)(ap);
        ua.u[1] = *(const v4u*)(ap + 16);
        const unsigned short* bp = Bl + (size_t)kt * 512;
        ub.u[0] = *(const v4u*)(bp);
        ub.u[1] = *(const v4u*)(bp + 8);
        if (kt + 1 < KT)
            __builtin_prefetch(Bl + (size_t)(kt + 1) * 512, 0, 1);
        acc = __builtin_amdgcn_wmma_f32_16x16x32_bf16(
                  false, ua.v, false, ub.v, (short)0, acc, false, false);
    }

    // C layout: VGPR i -> row (i + (lane<16?0:8)), col = lane&15 within tile
    const int crow0 = mtile * 16 + (lane < 16 ? 0 : 8);
    const int ccol  = ntile * 16 + (lane & 15);
    const float bv  = bias ? bias[ccol] : 0.f;
#pragma unroll
    for (int i = 0; i < 8; ++i) {
        float v = acc[i] + bv;
        if (act == 1) v = tanhf(v);
        size_t off = (size_t)(crow0 + i) * N + ccol;
        if (outF) outF[off] = v;
        if (outB) outB[off] = f2bf(v);
    }
}

// ---------------------------------------------------------------------
// factorized embedding: embBf[t,b,:] = bf16(E[token[b,t]] @ P), t<T-1
// ---------------------------------------------------------------------
__global__ void embed_kernel(const int* __restrict__ tgt,
                             const float* __restrict__ E,
                             const float* __restrict__ P,
                             unsigned short* __restrict__ embBf)
{
    const int tb = blockIdx.x;            // tb = t*B + b
    const int t = tb >> 5, b = tb & 31;
    const int tok = tgt[b * T_ + t];
    const float* Er = E + (size_t)tok * RANK_;
    for (int e = threadIdx.x; e < EMB_; e += blockDim.x) {
        float a = 0.f;
#pragma unroll 4
        for (int k = 0; k < RANK_; ++k) a += Er[k] * P[(size_t)k * EMB_ + e];
        embBf[(size_t)tb * EMB_ + e] = f2bf(a);
    }
}

// ---------------------------------------------------------------------
// state init: h0/h1 <- encoder_state, c0/c1/ctx <- 0
// ---------------------------------------------------------------------
__global__ void init_state_kernel(const float* __restrict__ es,
                                  float* h0, float* h1,
                                  float* c0, float* c1, float* ctx)
{
    int i = blockIdx.x * blockDim.x + threadIdx.x;   // 0 .. B*H-1
    h0[i] = es[i];
    h1[i] = es[B_ * H_ + i];
    c0[i] = 0.f; c1[i] = 0.f; ctx[i] = 0.f;
}

// x_cat0 = bf16([emb_t | ctx | h0_prev])  [B, 2560]
__global__ void build_xcat0_kernel(const unsigned short* __restrict__ embBf, int t,
                                   const float* __restrict__ ctx,
                                   const float* __restrict__ h0,
                                   unsigned short* __restrict__ x)
{
    int idx = blockIdx.x * blockDim.x + threadIdx.x;   // B*KX0
    int b = idx / KX0_, col = idx % KX0_;
    unsigned short v;
    if (col < EMB_)            v = embBf[((size_t)t * B_ + b) * EMB_ + col];
    else if (col < EMB_ + C_)  v = f2bf(ctx[b * C_ + (col - EMB_)]);
    else                       v = f2bf(h0[b * H_ + (col - EMB_ - C_)]);
    x[idx] = v;
}

// x_cat1 = bf16([h0_new | h1_prev])  [B, 2048]
__global__ void build_xcat1_kernel(const float* __restrict__ h0,
                                   const float* __restrict__ h1,
                                   unsigned short* __restrict__ x)
{
    int idx = blockIdx.x * blockDim.x + threadIdx.x;   // B*KX1
    int b = idx >> 11, col = idx & 2047;
    x[idx] = f2bf(col < H_ ? h0[b * H_ + col] : h1[b * H_ + (col - H_)]);
}

// torch LSTMCell gate order i,f,g,o along 4H; in-place h/c update
__global__ void lstm_cell_kernel(const float* __restrict__ gates,
                                 float* __restrict__ h, float* __restrict__ c)
{
    int idx = blockIdx.x * blockDim.x + threadIdx.x;   // B*H
    int b = idx >> 10, j = idx & 1023;
    const float* g = gates + (size_t)b * (4 * H_);
    float i_ = sigmoidf_(g[j]);
    float f_ = sigmoidf_(g[H_ + j]);
    float g_ = tanhf(g[2 * H_ + j]);
    float o_ = sigmoidf_(g[3 * H_ + j]);
    float cn = f_ * c[idx] + i_ * g_;
    c[idx] = cn;
    h[idx] = o_ * tanhf(cn);
}

// masked dot-product attention over encoder outputs; one block per batch b
__global__ void attention_kernel(const float* __restrict__ enc,   // [S,B,C]
                                 const int* __restrict__ enc_lens,
                                 const float* __restrict__ h1,    // [B,H]
                                 float* __restrict__ ctx)         // [B,C]
{
    const int b = blockIdx.x;
    __shared__ float sc[S_];
    __shared__ float sden;
    const int wave = threadIdx.x >> 5, lane = threadIdx.x & 31;

    for (int s = wave; s < S_; s += 8) {
        const float* er = enc + ((size_t)s * B_ + b) * C_;
        float p = 0.f;
        for (int k = lane; k < C_; k += 32) p += er[k] * h1[(size_t)b * H_ + k];
#pragma unroll
        for (int off = 16; off; off >>= 1) p += __shfl_xor(p, off, 32);
        if (lane == 0) sc[s] = p;
    }
    __syncthreads();

    if (threadIdx.x == 0) {
        int L = enc_lens[b];
        float mx = -1e30f;
        for (int s = 0; s < L; ++s) mx = fmaxf(mx, sc[s]);
        float sum = 0.f;
        for (int s = 0; s < S_; ++s) {
            float p = (s < L) ? __expf(sc[s] - mx) : 0.f;
            sc[s] = p; sum += p;
        }
        sden = 1.f / sum;
    }
    __syncthreads();

    for (int cix = threadIdx.x; cix < C_; cix += blockDim.x) {
        float a = 0.f;
#pragma unroll 8
        for (int s = 0; s < S_; ++s)
            a += sc[s] * enc[((size_t)s * B_ + b) * C_ + cix];
        ctx[(size_t)b * C_ + cix] = a * sden;
    }
}

// featsBf[t*B+b, :] = bf16([h1 | ctx])
__global__ void record_feats_kernel(int t, const float* __restrict__ h1,
                                    const float* __restrict__ ctx,
                                    unsigned short* __restrict__ featsBf)
{
    int idx = blockIdx.x * blockDim.x + threadIdx.x;   // B*KF
    int b = idx >> 11, col = idx & 2047;
    featsBf[((size_t)t * B_ + b) * KF_ + col] =
        f2bf(col < H_ ? h1[b * H_ + col] : ctx[b * C_ + (col - H_)]);
}

// ---------------------------------------------------------------------
// Fused vocab projection + online log-sum-exp cross-entropy.
// One block per output row m = t*B + b.  logits[v] = r[m,:] . E[v,:]
// Never materializes the 1504x32000 logits tensor.
// ---------------------------------------------------------------------
__global__ void ce_loss_kernel(const float* __restrict__ r,    // [MOUT,RANK]
                               const float* __restrict__ E,    // [VOCAB,RANK]
                               const int* __restrict__ tgt,    // [B,T]
                               const int* __restrict__ tgt_lens,
                               float* __restrict__ nll)        // [MOUT]
{
    const int m = blockIdx.x;
    const int t = m >> 5, b = m & 31;
    const bool valid = t < (tgt_lens[b] - 1);
    if (!valid) { if (threadIdx.x == 0) nll[m] = 0.f; return; }

    const int label = tgt[b * T_ + t + 1];
    __shared__ float rr[RANK_];
    __shared__ float smax[256], ssum[256], slbl;
    if (threadIdx.x < RANK_) rr[threadIdx.x] = r[(size_t)m * RANK_ + threadIdx.x];
    __syncthreads();

    float mx = -1e30f, sm = 0.f;
    for (int v = threadIdx.x; v < VOCAB_; v += 256) {
        const float* Ev = E + (size_t)v * RANK_;
        float d = 0.f;
#pragma unroll 8
        for (int k = 0; k < RANK_; ++k) d += rr[k] * Ev[k];
        if (d > mx) { sm = sm * __expf(mx - d) + 1.f; mx = d; }
        else        { sm += __expf(d - mx); }
        if (v == label) slbl = d;
    }
    smax[threadIdx.x] = mx; ssum[threadIdx.x] = sm;
    __syncthreads();
    for (int stride = 128; stride; stride >>= 1) {
        if (threadIdx.x < stride) {
            float m1 = smax[threadIdx.x], s1 = ssum[threadIdx.x];
            float m2 = smax[threadIdx.x + stride], s2 = ssum[threadIdx.x + stride];
            if (m1 >= m2) { smax[threadIdx.x] = m1; ssum[threadIdx.x] = s1 + s2 * __expf(m2 - m1); }
            else          { smax[threadIdx.x] = m2; ssum[threadIdx.x] = s2 + s1 * __expf(m1 - m2); }
        }
        __syncthreads();
    }
    if (threadIdx.x == 0)
        nll[m] = (smax[0] + logf(ssum[0])) - slbl;
}

// deterministic final reduction of per-row NLL into the scalar loss
__global__ void reduce_loss_kernel(const float* __restrict__ nll, float* __restrict__ out)
{
    __shared__ float ss[256];
    float a = 0.f;
    for (int i = threadIdx.x; i < MOUT_; i += 256) a += nll[i];
    ss[threadIdx.x] = a;
    __syncthreads();
    for (int stride = 128; stride; stride >>= 1) {
        if (threadIdx.x < stride) ss[threadIdx.x] += ss[threadIdx.x + stride];
        __syncthreads();
    }
    if (threadIdx.x == 0) out[0] = ss[0];
}

// ---------------------------------------------------------------------
extern "C" void kernel_launch(void* const* d_in, const int* in_sizes, int n_in,
                              void* d_out, int out_size, void* d_ws, size_t ws_size,
                              hipStream_t stream)
{
    const float* encoded = (const float*)d_in[0];   // [S,B,C]
    const float* encst   = (const float*)d_in[1];   // [2,B,H]
    const int*   tgt     = (const int*)d_in[2];     // [B,T]
    const int*   enclen  = (const int*)d_in[3];     // [B]
    const int*   tgtlen  = (const int*)d_in[4];     // [B]
    const float* E  = (const float*)d_in[5];        // [VOCAB,RANK]
    const float* P  = (const float*)d_in[6];        // [RANK,EMB]
    const float* W0 = (const float*)d_in[7];        // [EMB+C,4H]
    const float* U0 = (const float*)d_in[8];        // [H,4H]
    const float* b0 = (const float*)d_in[9];
    const float* W1 = (const float*)d_in[10];
    const float* U1 = (const float*)d_in[11];
    const float* b1 = (const float*)d_in[12];
    const float* Wo = (const float*)d_in[13];       // [H+C,EMB]
    const float* bo = (const float*)d_in[14];
    float* out = (float*)d_out;

    // ---- carve workspace (256B aligned) ----
    char* wsp = (char*)d_ws;
    size_t off = 0;
    auto carve = [&](size_t bytes) -> void* {
        void* p = wsp + off;
        off += (bytes + 255) & ~(size_t)255;
        return p;
    };
    unsigned short* pw01   = (unsigned short*)carve((size_t)(4 * H_) * KX0_ * 2); // W0;U0 packed
    unsigned short* pw11   = (unsigned short*)carve((size_t)(4 * H_) * KX1_ * 2); // W1;U1 packed
    unsigned short* pwo    = (unsigned short*)carve((size_t)EMB_ * KF_ * 2);      // Wo packed
    unsigned short* ppt    = (unsigned short*)carve((size_t)RANK_ * EMB_ * 2);    // P^T packed
    unsigned short* embBf  = (unsigned short*)carve((size_t)MOUT_ * EMB_ * 2);
    unsigned short* xcat0  = (unsigned short*)carve((size_t)B_ * KX0_ * 2);
    unsigned short* xcat1  = (unsigned short*)carve((size_t)B_ * KX1_ * 2);
    float* gates  = (float*)carve((size_t)B_ * 4 * H_ * 4);
    float* h0     = (float*)carve((size_t)B_ * H_ * 4);
    float* c0     = (float*)carve((size_t)B_ * H_ * 4);
    float* h1     = (float*)carve((size_t)B_ * H_ * 4);
    float* c1     = (float*)carve((size_t)B_ * H_ * 4);
    float* ctx    = (float*)carve((size_t)B_ * C_ * 4);
    unsigned short* featsBf = (unsigned short*)carve((size_t)MOUT_ * KF_ * 2);
    unsigned short* hprojBf = (unsigned short*)carve((size_t)MOUT_ * EMB_ * 2);
    float* rbuf   = (float*)carve((size_t)MOUT_ * RANK_ * 4);
    float* nllbuf = (float*)carve((size_t)MOUT_ * 4);

    // ---- one-time-per-call prep ----
    init_state_kernel<<<(B_ * H_) / 256, 256, 0, stream>>>(encst, h0, h1, c0, c1, ctx);

    auto pack = [&](const float* src, unsigned short* dst, int N, int K,
                    int KT_total, int kt_off, int transposed) {
        long total = (long)(N >> 4) * (K >> 5) * 512;
        pack_b_kernel<<<(unsigned)((total + 255) / 256), 256, 0, stream>>>(
            src, dst, N, K, KT_total, kt_off, transposed);
    };
    pack(W0, pw01, 4 * H_, EMB_ + C_, KX0_ / 32, 0, 0);            // K rows 0..1535
    pack(U0, pw01, 4 * H_, H_,        KX0_ / 32, (EMB_ + C_) / 32, 0);
    pack(W1, pw11, 4 * H_, H_,        KX1_ / 32, 0, 0);
    pack(U1, pw11, 4 * H_, H_,        KX1_ / 32, H_ / 32, 0);
    pack(Wo, pwo, EMB_, KF_, KF_ / 32, 0, 0);
    pack(P,  ppt, RANK_, EMB_, EMB_ / 32, 0, 1);                   // P^T via transpose

    embed_kernel<<<MOUT_, 256, 0, stream>>>(tgt, E, P, embBf);

    // ---- sequential decode: 47 steps ----
    for (int t = 0; t < T_ - 1; ++t) {
        build_xcat0_kernel<<<(B_ * KX0_) / 256, 256, 0, stream>>>(embBf, t, ctx, h0, xcat0);
        wmma_gemm_kernel<<<dim3(4 * H_ / 64, B_ / 16), 128, 0, stream>>>(
            xcat0, pw01, b0, gates, nullptr, B_, 4 * H_, KX0_, 0);
        lstm_cell_kernel<<<(B_ * H_) / 256, 256, 0, stream>>>(gates, h0, c0);

        build_xcat1_kernel<<<(B_ * KX1_) / 256, 256, 0, stream>>>(h0, h1, xcat1);
        wmma_gemm_kernel<<<dim3(4 * H_ / 64, B_ / 16), 128, 0, stream>>>(
            xcat1, pw11, b1, gates, nullptr, B_, 4 * H_, KX1_, 0);
        lstm_cell_kernel<<<(B_ * H_) / 256, 256, 0, stream>>>(gates, h1, c1);

        attention_kernel<<<B_, 256, 0, stream>>>(encoded, enclen, h1, ctx);
        record_feats_kernel<<<(B_ * KF_) / 256, 256, 0, stream>>>(t, h1, ctx, featsBf);
    }

    // ---- output head ----
    // hproj = tanh(feats @ Wo + bo)  -> bf16 for next GEMM
    wmma_gemm_kernel<<<dim3(EMB_ / 64, MOUT_ / 16), 128, 0, stream>>>(
        featsBf, pwo, bo, nullptr, hprojBf, MOUT_, EMB_, KF_, 1);
    // r = hproj @ P^T  (f32, consumed by fused CE)
    wmma_gemm_kernel<<<dim3(RANK_ / 64, MOUT_ / 16), 128, 0, stream>>>(
        hprojBf, ppt, nullptr, rbuf, nullptr, MOUT_, RANK_, EMB_, 0);

    // fused vocab matmul + log-softmax + NLL (streaming, no logits tensor)
    ce_loss_kernel<<<MOUT_, 256, 0, stream>>>(rbuf, E, tgt, tgtlen, nllbuf);
    reduce_loss_kernel<<<1, 256, 0, stream>>>(nllbuf, out);
}
